// OptimizedxLSTM_14078902796644
// MI455X (gfx1250) — matrix operand order repaired and checked by reading, not compiled, not run on hardware
//
#include <hip/hip_runtime.h>

// ---------------------------------------------------------------------------
// xLSTM forward for MI455X (gfx1250, wave32, WMMA).
// Scan restructured: all projections batched over B*T=512 rows (recurrence is
// state-only); two sequential kernels (mLSTM cell in LDS, sLSTM rec-GEMM loop).
// bf16 WMMA f32-accumulate for GEMMs; f32 VALU for gates/norms.
// Head GEMM streams weight tiles via the Tensor Data Mover, double-buffered.
// ---------------------------------------------------------------------------

#define Bsz    8
#define Tsz    64
#define Hdim   1024
#define NHh    16
#define HSd    128
#define INNERd 2048
#define QKV3   6144
#define FOURH  4096
#define Vvoc   32000
#define KCONV  4
#define ROWS   (Bsz * Tsz)          // 512

typedef __bf16 bf16_t;
typedef __attribute__((ext_vector_type(16))) __bf16        v16bf;
typedef __attribute__((ext_vector_type(8)))  __bf16        v8bf;
typedef __attribute__((ext_vector_type(8)))  float         v8f;
typedef __attribute__((ext_vector_type(4)))  unsigned int  u32x4;
typedef __attribute__((ext_vector_type(8)))  int           i32x8;
typedef __attribute__((ext_vector_type(4)))  int           i32x4;

// ----------------------------- device helpers ------------------------------

__device__ __forceinline__ bf16_t f2bf(float f) {
  unsigned u = __builtin_bit_cast(unsigned, f);
  unsigned r = (u + 0x7FFFu + ((u >> 16) & 1u)) >> 16;
  unsigned short s = (unsigned short)r;
  return __builtin_bit_cast(bf16_t, s);
}
__device__ __forceinline__ float logsigf(float x) {
  return fminf(x, 0.f) - log1pf(expf(-fabsf(x)));
}
__device__ __forceinline__ float softcapf(float x) { return 30.f * tanhf(x * (1.f/30.f)); }
__device__ __forceinline__ float siluf(float x)    { return x / (1.f + expf(-x)); }

__device__ __forceinline__ float block_reduce_sum(float v, float* red) {
  int t = threadIdx.x;
  red[t] = v; __syncthreads();
  for (int s = blockDim.x >> 1; s > 0; s >>= 1) {
    if (t < s) red[t] += red[t + s];
    __syncthreads();
  }
  float r = red[0]; __syncthreads();
  return r;
}

// WMMA fragment loaders (wave32; contiguous k = 16*(lane/16)+s in both A and B, not the ISA table order).
// A (16x32 bf16, row-major M x K): lanes 0-15 hold M=lane, K=[0..15];
// lanes 16-31 hold M=lane-16, K=[16..31].
__device__ __forceinline__ v16bf load_a_frag_g(const bf16_t* A, long lda, int m0, int k0) {
  int l   = threadIdx.x & 31;
  int seg = l >> 4;
  const bf16_t* p = A + (long)(m0 + (l & 15)) * lda + k0 + seg * 16;
  return *(const v16bf*)p;
}
// B (32x16 bf16, weights stored N x K row-major): lanes 0-15: n=lane, K=[0..15];
// lanes 16-31: n=lane-16, K=[16..31].  One contiguous 32B load.
// Row index clamped to [0, nmax) so padding sub-tiles never read out of bounds.
__device__ __forceinline__ v16bf load_b_frag_g(const bf16_t* Bw, long ldb, int n0, int k0,
                                               int nmax) {
  int l = threadIdx.x & 31;
  int row = n0 + (l & 15);
  row = row < nmax ? row : nmax - 1;
  const bf16_t* p = Bw + (long)row * ldb + k0 + ((l >> 4) << 4);
  return *(const v16bf*)p;
}
__device__ __forceinline__ v16bf load_b_frag_lds(const bf16_t* tile, int nloc, int kk) {
  int l = threadIdx.x & 31;
  const bf16_t* p = tile + (nloc + (l & 15)) * 64 + kk + ((l >> 4) << 4);
  return *(const v16bf*)p;
}

// ------------------------------- kernels -----------------------------------

__global__ void k_f32_to_bf16(const float* __restrict__ s, bf16_t* __restrict__ d, long n) {
  long i = (long)blockIdx.x * blockDim.x + threadIdx.x;
  long stride = (long)gridDim.x * blockDim.x;
  for (; i < n; i += stride) d[i] = f2bf(s[i]);
}

__global__ void k_embed(const int* __restrict__ tok, const float* __restrict__ emb,
                        float* __restrict__ X0) {
  long row = blockIdx.x;
  long base = (long)tok[row] * Hdim;
  for (int c = threadIdx.x; c < Hdim; c += blockDim.x)
    X0[row * Hdim + c] = emb[base + c];
}

__global__ void k_init(bf16_t* rec_bf, float* c_st, float* n_st, float* m_st) {
  int i = blockIdx.x * blockDim.x + threadIdx.x;
  if (i < 16 * Hdim) rec_bf[i] = f2bf(0.f);
  if (i < Bsz * Hdim) { c_st[i] = 0.f; n_st[i] = 0.f; m_st[i] = -__builtin_inff(); }
}

// RMS norm: one block per row (256 threads), cols=Hdim.
__global__ void k_rmsnorm(const float* __restrict__ X, const float* __restrict__ w,
                          float* __restrict__ outf, bf16_t* __restrict__ outbf, int cols) {
  __shared__ float red[256];
  long row = blockIdx.x;
  const float* x = X + row * cols;
  float ss = 0.f;
  for (int c = threadIdx.x; c < cols; c += blockDim.x) { float v = x[c]; ss += v * v; }
  ss = block_reduce_sum(ss, red);
  float scale = rsqrtf(ss / (float)cols + 1e-6f);
  for (int c = threadIdx.x; c < cols; c += blockDim.x) {
    float v = x[c] * scale * w[c];
    if (outf)  outf[row * cols + c]  = v;
    if (outbf) outbf[row * cols + c] = f2bf(v);
  }
}

// Generic bf16 WMMA GEMM: C[M,N] = A[M,K] @ Bw[N,K]^T (+bias)(+resid)(softcap).
// grid = (ceil(N/256), Mpad/16), block = 128 (4 waves).  Each wave owns a
// 16x64 output tile: one A fragment is reused across 4 B fragments per k-step
// (4 independent WMMA accumulation chains hide L2 latency; A traffic /4).
__global__ void k_gemm(const bf16_t* __restrict__ A, const bf16_t* __restrict__ Bw,
                       const float* __restrict__ bias, const float* __restrict__ resid,
                       long resid_stride,
                       float* __restrict__ C, bf16_t* __restrict__ Cbf,
                       int Mreal, int N, int K, int do_softcap) {
  int wave = threadIdx.x >> 5;
  int nw = blockIdx.x * 256 + wave * 64;
  int m0 = blockIdx.y * 16;
  if (nw >= N) return;
  v8f acc[4] = { {}, {}, {}, {} };
  for (int k0 = 0; k0 < K; k0 += 32) {
    int l = threadIdx.x & 31;
    __builtin_prefetch(Bw + (long)(nw + (l & 15)) * K + k0 + 64, 0, 3);
    v16bf a = load_a_frag_g(A, K, m0, k0);
#pragma unroll
    for (int j = 0; j < 4; ++j) {
      v16bf b = load_b_frag_g(Bw, K, nw + j * 16, k0, N);
      acc[j] = __builtin_amdgcn_wmma_f32_16x16x32_bf16(false, a, false, b, (short)0,
                                                       acc[j], false, false);
    }
  }
  int l = threadIdx.x & 31;
  int mbase = m0 + ((l >> 4) << 3);
#pragma unroll
  for (int j = 0; j < 4; ++j) {
    int nn = nw + j * 16 + (l & 15);
    if (nn >= N) continue;
    float bv = bias ? bias[nn] : 0.f;
#pragma unroll
    for (int r = 0; r < 8; ++r) {
      int m = mbase + r;
      if (m < Mreal) {
        float v = acc[j][r] + bv;
        if (resid) v += resid[(long)m * resid_stride + nn];
        if (do_softcap) v = softcapf(v);
        long o = (long)m * N + nn;
        C[o] = v;
        if (Cbf) Cbf[o] = f2bf(v);
      }
    }
  }
}

// Depthwise causal conv (K=4, zero initial state) + SiLU, batched over time.
__global__ void k_conv_silu(const float* __restrict__ xin, const float* __restrict__ w,
                            const float* __restrict__ b, float* __restrict__ outf,
                            bf16_t* __restrict__ outbf, int C) {
  long i = (long)blockIdx.x * blockDim.x + threadIdx.x;
  long total = (long)ROWS * C;
  if (i >= total) return;
  int c = (int)(i % C);
  long row = i / C;
  int t = (int)(row % Tsz);
  float acc = b[c];
#pragma unroll
  for (int j = 0; j < KCONV; ++j) {
    int dt = KCONV - 1 - j;                  // w[c,j] multiplies x[t-(K-1-j)]
    if (t - dt >= 0) acc += w[c * KCONV + j] * xin[(row - dt) * C + c];
  }
  float s = siluf(acc);
  outf[i] = s;
  if (outbf) outbf[i] = f2bf(s);
}

// Sequential mLSTM cell: block per (b, head, half). 64 threads own 64 columns
// of the 128x128 cell state held in LDS across all 64 timesteps.
__global__ void k_mlstm_seq(const float* __restrict__ qkvf, const float* __restrict__ igf,
                            const float* __restrict__ fgf, float* __restrict__ mout) {
  __shared__ float cell[HSd * 64];
  __shared__ float kls[HSd];
  __shared__ float qls[HSd];
  __shared__ float red[64];
  __shared__ float scal[4];
  int half = blockIdx.x & 1;
  int h = (blockIdx.x >> 1) & (NHh - 1);
  int b = blockIdx.x >> 5;
  int tid = threadIdx.x;
  for (int d = 0; d < HSd; ++d) cell[d * 64 + tid] = 0.f;
  float nrm0 = 0.f, nrm1 = 0.f;
  float mprev = 0.f;                           // reference init: mx = zeros
  const float kscale = 0.08838834764831845f;   // 1/sqrt(128)
  for (int t = 0; t < Tsz; ++t) {
    long row = (long)b * Tsz + t;
    const float* qp = qkvf + row * QKV3 + h * HSd;
    const float* kp = qp + INNERd;
    const float* vp = qp + 2 * INNERd;
    qls[tid] = qp[tid];            qls[tid + 64] = qp[tid + 64];
    kls[tid] = kp[tid] * kscale;   kls[tid + 64] = kp[tid + 64] * kscale;
    float v_e = vp[half * 64 + tid];
    if (tid == 0) {
      float igv = igf[row * NHh + h];
      float fgv = fgf[row * NHh + h];
      float lf = logsigf(fgv);
      float mnew = fmaxf(igv, mprev + lf);
      scal[0] = expf(igv - mnew);
      scal[1] = expf(lf + mprev - mnew);
      scal[2] = expf(-mnew);
      mprev = mnew;
    }
    __syncthreads();
    float i_g = scal[0], f_g = scal[1];
    float num = 0.f;
#pragma unroll 4
    for (int d = 0; d < HSd; ++d) {
      float cv = f_g * cell[d * 64 + tid] + i_g * kls[d] * v_e;
      cell[d * 64 + tid] = cv;
      num += qls[d] * cv;
    }
    nrm0 = f_g * nrm0 + i_g * kls[tid];
    nrm1 = f_g * nrm1 + i_g * kls[tid + 64];
    float qn = block_reduce_sum(qls[tid] * nrm0 + qls[tid + 64] * nrm1, red);
    if (tid == 0) scal[3] = fmaxf(fabsf(qn), scal[2]) + 1e-6f;
    __syncthreads();
    mout[row * INNERd + h * HSd + half * 64 + tid] = num / scal[3];
    __syncthreads();
  }
}

// mLSTM epilogue: GroupNorm(16 groups of 128) + (gn + skip*xc)*silu(xg) -> bf16.
__global__ void k_gn_mix(const float* __restrict__ mout, const float* __restrict__ gn_g,
                         const float* __restrict__ gn_b, const float* __restrict__ skipw,
                         const float* __restrict__ xc, const float* __restrict__ xg,
                         bf16_t* __restrict__ h_bf) {
  __shared__ float red[128];
  int row = blockIdx.x >> 4;
  int grp = blockIdx.x & 15;
  int c = grp * HSd + threadIdx.x;
  long idx = (long)row * INNERd + c;
  float v = mout[idx];
  float mu = block_reduce_sum(v, red) * (1.f / (float)HSd);
  float d = v - mu;
  float var = block_reduce_sum(d * d, red) * (1.f / (float)HSd);
  float xn = d * rsqrtf(var + 1e-5f);
  float o = xn * gn_g[c] + gn_b[c];
  float g = xg[idx];
  h_bf[idx] = f2bf((o + skipw[c] * xc[idx]) * siluf(g));
}

// sLSTM pointwise + GroupNorm(16 groups of 64) + residual, one timestep.
__global__ void k_slstm_point(const float* __restrict__ g, float* __restrict__ c_st,
                              float* __restrict__ n_st, float* __restrict__ m_st,
                              bf16_t* __restrict__ rec_bf, const float* __restrict__ gn_g,
                              const float* __restrict__ gn_b, const float* __restrict__ Y1,
                              float* __restrict__ ys, int t) {
  __shared__ float red[64];
  int b = blockIdx.x >> 4;
  int grp = blockIdx.x & 15;
  int ch = grp * 64 + threadIdx.x;
  long gi = (long)b * FOURH;
  float iv = softcapf(g[gi + ch]);
  float fv = softcapf(g[gi + Hdim + ch]);
  float zv = softcapf(g[gi + 2 * Hdim + ch]);
  float ov = softcapf(g[gi + 3 * Hdim + ch]);
  long si = (long)b * Hdim + ch;
  float m = m_st[si];
  float lfm = m + logsigf(fv);
  float mnew = fmaxf(iv, lfm);
  float og = 1.f / (1.f + expf(-ov));
  float ii = expf(iv - mnew);
  float ff = expf(lfm - mnew);
  float c = ff * c_st[si] + ii * tanhf(zv);
  float n = ff * n_st[si] + ii;
  float y = og * c / (n + 1e-6f);
  c_st[si] = c; n_st[si] = n; m_st[si] = mnew; rec_bf[si] = f2bf(y);
  float mu = block_reduce_sum(y, red) * (1.f / 64.f);
  float d = y - mu;
  float var = block_reduce_sum(d * d, red) * (1.f / 64.f);
  float xn = d * rsqrtf(var + 1e-5f);
  long row = (long)b * Tsz + t;
  ys[row * Hdim + ch] = xn * gn_g[ch] + gn_b[ch] + Y1[row * Hdim + ch];
}

// Head GEMM: logits[512, 32000] = ysn_bf @ head_bf^T.
// B-tiles (64n x 64k bf16, 8KB) double-buffered in LDS and streamed by the
// Tensor Data Mover: tile k+1 is issued before computing tile k, with
// s_wait_tensorcnt(1) keeping one transfer in flight behind the WMMA work.
// grid = (500, 8), block = 128; wave w owns m-rows [w*16, w*16+16).
__global__ void k_head_gemm(const bf16_t* __restrict__ A, const bf16_t* __restrict__ Bw,
                            float* __restrict__ C) {
  __shared__ __align__(32) bf16_t btile[2][64 * 64];   // assumed at LDS offset 0
  int wave = threadIdx.x >> 5;
  int m0 = blockIdx.y * 64 + wave * 16;
  int n0blk = blockIdx.x * 64;
  const int nK = Hdim / 64;
  v8f acc[4] = { {}, {}, {}, {} };

#if __has_builtin(__builtin_amdgcn_tensor_load_to_lds)
  auto tdm_issue = [&](int buf, int k0) {
    unsigned long long ga =
        (unsigned long long)(const void*)(Bw + (long)n0blk * Hdim + k0);
    u32x4 g0; i32x8 g1; i32x4 g2; i32x4 g3;
    // D# group0: count=1, lds_addr=buf*8KB, 57b global addr, type=2 (image).
    g0[0] = 1u;
    g0[1] = (unsigned)(buf * (64 * 64 * 2));
    g0[2] = (unsigned)(ga & 0xffffffffull);
    g0[3] = (unsigned)((ga >> 32) & 0x1ffffffull) | (2u << 30);
    // D# group1: data_size=2B, tensor 64x64, tile 64x64, dim0 stride=Hdim.
    g1[0] = (int)(1u << 16);
    g1[1] = (int)(64u << 16);        // tensor_dim0 lo16 @ bits[63:48]
    g1[2] = (int)(64u << 16);        // dim0 hi16 (0) | tensor_dim1 lo16
    g1[3] = (int)(64u << 16);        // dim1 hi16 (0) | tile_dim0=64
    g1[4] = (int)64;                 // tile_dim1=64, tile_dim2=0
    g1[5] = (int)Hdim;               // tensor_dim0_stride lo32
    g1[6] = 0; g1[7] = 0;
    g2[0] = g2[1] = g2[2] = g2[3] = 0;
    g3[0] = g3[1] = g3[2] = g3[3] = 0;
#if __clang_major__ >= 23
    i32x8 g4; g4[0]=g4[1]=g4[2]=g4[3]=g4[4]=g4[5]=g4[6]=g4[7]=0;
    __builtin_amdgcn_tensor_load_to_lds(g0, g1, g2, g3, g4, 0);
#else
    __builtin_amdgcn_tensor_load_to_lds(g0, g1, g2, g3, 0);
#endif
  };
  if (threadIdx.x < 32) tdm_issue(0, 0);
#endif

  for (int ki = 0; ki < nK; ++ki) {
    int k0 = ki * 64;
#if __has_builtin(__builtin_amdgcn_tensor_load_to_lds)
    if (threadIdx.x < 32) {
      if (ki + 1 < nK) {
        tdm_issue((ki + 1) & 1, k0 + 64);          // prefetch next tile
#if __has_builtin(__builtin_amdgcn_s_wait_tensorcnt)
        __builtin_amdgcn_s_wait_tensorcnt(1);      // current tile complete
#else
        asm volatile("s_wait_tensorcnt 1" ::: "memory");
#endif
      } else {
#if __has_builtin(__builtin_amdgcn_s_wait_tensorcnt)
        __builtin_amdgcn_s_wait_tensorcnt(0);
#else
        asm volatile("s_wait_tensorcnt 0" ::: "memory");
#endif
      }
    }
#else
    for (int idx = threadIdx.x; idx < (64 * 64) / 8; idx += 128) {
      int rrow = idx >> 3, c8 = (idx & 7) << 3;
      *(v8bf*)(&btile[ki & 1][0] + rrow * 64 + c8) =
          *(const v8bf*)(Bw + (long)(n0blk + rrow) * Hdim + k0 + c8);
    }
#endif
    __syncthreads();
    const bf16_t* tile = &btile[ki & 1][0];
#pragma unroll
    for (int kk = 0; kk < 64; kk += 32) {
      v16bf a = load_a_frag_g(A, Hdim, m0, k0 + kk);
#pragma unroll
      for (int j = 0; j < 4; ++j) {
        v16bf b = load_b_frag_lds(tile, j * 16, kk);
        acc[j] = __builtin_amdgcn_wmma_f32_16x16x32_bf16(false, a, false, b, (short)0,
                                                         acc[j], false, false);
      }
    }
    __syncthreads();
  }
  int l = threadIdx.x & 31;
  int mbase = m0 + ((l >> 4) << 3);
#pragma unroll
  for (int j = 0; j < 4; ++j) {
    int nn = n0blk + j * 16 + (l & 15);
#pragma unroll
    for (int r = 0; r < 8; ++r)
      C[(long)(mbase + r) * Vvoc + nn] = acc[j][r];
  }
}

// ------------------------------ host driver --------------------------------

extern "C" void kernel_launch(void* const* d_in, const int* in_sizes, int n_in,
                              void* d_out, int out_size, void* d_ws, size_t ws_size,
                              hipStream_t stream) {
  (void)in_sizes; (void)n_in; (void)out_size; (void)ws_size;
  // Input order: tokens, then params flattened leaf-wise in sorted-key order.
  const int*   tokens   = (const int*)  d_in[0];
  const float* emb      = (const float*)d_in[1];
  const float* final_w  = (const float*)d_in[2];
  const float* head_w   = (const float*)d_in[3];
  const float* m_conv_b = (const float*)d_in[4];
  const float* m_conv_w = (const float*)d_in[5];
  const float* m_down   = (const float*)d_in[6];
  const float* m_fg_b   = (const float*)d_in[7];
  const float* m_fg_w   = (const float*)d_in[8];
  const float* m_gate   = (const float*)d_in[9];
  const float* m_gn_b   = (const float*)d_in[10];
  const float* m_gn_g   = (const float*)d_in[11];
  const float* m_ig_b   = (const float*)d_in[12];
  const float* m_ig_w   = (const float*)d_in[13];
  const float* m_qkvw   = (const float*)d_in[14];
  const float* m_rms_w  = (const float*)d_in[15];
  const float* m_skip   = (const float*)d_in[16];
  const float* m_xproj  = (const float*)d_in[17];
  const float* s_conv_b = (const float*)d_in[18];
  const float* s_conv_w = (const float*)d_in[19];
  const float* s_gn_b   = (const float*)d_in[20];
  const float* s_gn_g   = (const float*)d_in[21];
  const float* s_in_w   = (const float*)d_in[22];
  const float* s_rms_w  = (const float*)d_in[23];
  const float* s_st_b   = (const float*)d_in[24];
  const float* s_st_w   = (const float*)d_in[25];
  float* logits = (float*)d_out;

  char* wp = (char*)d_ws;
  auto alloc = [&](size_t bytes) -> void* {
    void* r = (void*)wp;
    wp += (bytes + 255) & ~(size_t)255;
    return r;
  };

  // bf16 weight mirrors (L2-resident across the whole pass).
  bf16_t* head_bf  = (bf16_t*)alloc((size_t)Vvoc * Hdim * 2);
  bf16_t* xproj_bf = (bf16_t*)alloc((size_t)INNERd * Hdim * 2);
  bf16_t* gate_bf  = (bf16_t*)alloc((size_t)INNERd * Hdim * 2);
  bf16_t* qkvw_bf  = (bf16_t*)alloc((size_t)QKV3 * INNERd * 2);
  bf16_t* igw_bf   = (bf16_t*)alloc((size_t)NHh * QKV3 * 2);
  bf16_t* fgw_bf   = (bf16_t*)alloc((size_t)NHh * QKV3 * 2);
  bf16_t* downw_bf = (bf16_t*)alloc((size_t)Hdim * INNERd * 2);
  bf16_t* inw_bf   = (bf16_t*)alloc((size_t)FOURH * Hdim * 2);
  bf16_t* stw_bf   = (bf16_t*)alloc((size_t)FOURH * Hdim * 2);
  // Activations.
  float*  X0     = (float*) alloc((size_t)ROWS * Hdim * 4);
  bf16_t* xn_bf  = (bf16_t*)alloc((size_t)ROWS * Hdim * 2);
  float*  xm     = (float*) alloc((size_t)ROWS * INNERd * 4);
  float*  xg     = (float*) alloc((size_t)ROWS * INNERd * 4);
  float*  xc     = (float*) alloc((size_t)ROWS * INNERd * 4);
  bf16_t* xc_bf  = (bf16_t*)alloc((size_t)ROWS * INNERd * 2);
  float*  qkvf   = (float*) alloc((size_t)ROWS * QKV3 * 4);
  bf16_t* qkv_bf = (bf16_t*)alloc((size_t)ROWS * QKV3 * 2);
  float*  igf    = (float*) alloc((size_t)ROWS * NHh * 4);
  float*  fgf    = (float*) alloc((size_t)ROWS * NHh * 4);
  float*  moutb  = (float*) alloc((size_t)ROWS * INNERd * 4);
  bf16_t* h_bf   = (bf16_t*)alloc((size_t)ROWS * INNERd * 2);
  float*  Y1     = (float*) alloc((size_t)ROWS * Hdim * 4);
  float*  xns    = (float*) alloc((size_t)ROWS * Hdim * 4);
  float*  xcs    = (float*) alloc((size_t)ROWS * Hdim * 4);
  bf16_t* xcs_bf = (bf16_t*)alloc((size_t)ROWS * Hdim * 2);
  float*  pre    = (float*) alloc((size_t)ROWS * FOURH * 4);
  float*  gbuf   = (float*) alloc((size_t)16 * FOURH * 4);
  bf16_t* rec_bf = (bf16_t*)alloc((size_t)16 * Hdim * 2);
  float*  c_st   = (float*) alloc((size_t)Bsz * Hdim * 4);
  float*  n_st   = (float*) alloc((size_t)Bsz * Hdim * 4);
  float*  mst    = (float*) alloc((size_t)Bsz * Hdim * 4);
  float*  ys     = (float*) alloc((size_t)ROWS * Hdim * 4);
  bf16_t* ysn_bf = (bf16_t*)alloc((size_t)ROWS * Hdim * 2);

  // ---- weight conversion f32 -> bf16 ----
  struct Cv { const float* s; bf16_t* d; long n; } cv[] = {
    { head_w,  head_bf,  (long)Vvoc * Hdim },
    { m_xproj, xproj_bf, (long)INNERd * Hdim },
    { m_gate,  gate_bf,  (long)INNERd * Hdim },
    { m_qkvw,  qkvw_bf,  (long)QKV3 * INNERd },
    { m_ig_w,  igw_bf,   (long)NHh * QKV3 },
    { m_fg_w,  fgw_bf,   (long)NHh * QKV3 },
    { m_down,  downw_bf, (long)Hdim * INNERd },
    { s_in_w,  inw_bf,   (long)FOURH * Hdim },
    { s_st_w,  stw_bf,   (long)FOURH * Hdim },
  };
  for (auto& c : cv) {
    int blocks = (int)((c.n + 255) / 256);
    k_f32_to_bf16<<<blocks, 256, 0, stream>>>(c.s, c.d, c.n);
  }
  k_init<<<(16 * Hdim + 255) / 256, 256, 0, stream>>>(rec_bf, c_st, n_st, mst);

  // ---- batched (time-parallel) pipeline ----
  k_embed<<<ROWS, 256, 0, stream>>>(tokens, emb, X0);
  k_rmsnorm<<<ROWS, 256, 0, stream>>>(X0, m_rms_w, nullptr, xn_bf, Hdim);
  k_gemm<<<dim3(INNERd / 256, ROWS / 16), 128, 0, stream>>>(
      xn_bf, xproj_bf, nullptr, nullptr, 0, xm, nullptr, ROWS, INNERd, Hdim, 0);
  k_gemm<<<dim3(INNERd / 256, ROWS / 16), 128, 0, stream>>>(
      xn_bf, gate_bf, nullptr, nullptr, 0, xg, nullptr, ROWS, INNERd, Hdim, 0);
  k_conv_silu<<<(ROWS * INNERd) / 256, 256, 0, stream>>>(xm, m_conv_w, m_conv_b,
                                                         xc, xc_bf, INNERd);
  k_gemm<<<dim3(QKV3 / 256, ROWS / 16), 128, 0, stream>>>(
      xc_bf, qkvw_bf, nullptr, nullptr, 0, qkvf, qkv_bf, ROWS, QKV3, INNERd, 0);
  k_gemm<<<dim3(1, ROWS / 16), 128, 0, stream>>>(
      qkv_bf, igw_bf, m_ig_b, nullptr, 0, igf, nullptr, ROWS, NHh, QKV3, 1);
  k_gemm<<<dim3(1, ROWS / 16), 128, 0, stream>>>(
      qkv_bf, fgw_bf, m_fg_b, nullptr, 0, fgf, nullptr, ROWS, NHh, QKV3, 1);
  k_mlstm_seq<<<Bsz * NHh * 2, 64, 0, stream>>>(qkvf, igf, fgf, moutb);
  k_gn_mix<<<ROWS * NHh, 128, 0, stream>>>(moutb, m_gn_g, m_gn_b, m_skip, xc, xg, h_bf);
  k_gemm<<<dim3(Hdim / 256, ROWS / 16), 128, 0, stream>>>(
      h_bf, downw_bf, nullptr, X0, Hdim, Y1, nullptr, ROWS, Hdim, INNERd, 0);

  // ---- sLSTM ----
  k_rmsnorm<<<ROWS, 256, 0, stream>>>(Y1, s_rms_w, xns, nullptr, Hdim);
  k_conv_silu<<<(ROWS * Hdim) / 256, 256, 0, stream>>>(xns, s_conv_w, s_conv_b,
                                                       xcs, xcs_bf, Hdim);
  k_gemm<<<dim3(FOURH / 256, ROWS / 16), 128, 0, stream>>>(
      xcs_bf, inw_bf, s_st_b, nullptr, 0, pre, nullptr, ROWS, FOURH, Hdim, 0);
  for (int t = 0; t < Tsz; ++t) {
    k_gemm<<<dim3(FOURH / 256, 1), 128, 0, stream>>>(
        rec_bf, stw_bf, nullptr, pre + (size_t)t * FOURH, (long)Tsz * FOURH,
        gbuf, nullptr, Bsz, FOURH, Hdim, 0);
    k_slstm_point<<<Bsz * NHh, 64, 0, stream>>>(gbuf, c_st, n_st, mst, rec_bf,
                                                s_gn_g, s_gn_b, Y1, ys, t);
  }

  // ---- head ----
  k_rmsnorm<<<ROWS, 256, 0, stream>>>(ys, final_w, nullptr, ysn_bf, Hdim);
  k_head_gemm<<<dim3(Vvoc / 64, ROWS / 64), 128, 0, stream>>>(ysn_bf, head_bf, logits);
}
